// attention_kernel_87986700026103
// MI455X (gfx1250) — compile-verified, matrix operand order checked by reading, not run on hardware
//
#include <hip/hip_runtime.h>

// ---------------------------------------------------------------------------
// FAVOR+ causal linear attention for MI455X (gfx1250), bf16 WMMA pipeline.
// H=16 heads, L=4096, D=64, R=128 features, CHUNK=128, nc=32 chunks.
// v is augmented with a ones column (DP=80) so numerator and denominator
// share the same WMMA chains. Round 2: native bf16 cvt + async global->LDS.
// ---------------------------------------------------------------------------

#define H_   16
#define L_   4096
#define D_   64
#define R_   128
#define C_   128          // chunk
#define NC_  32           // L_/C_
#define DP_  80           // D_ + 16 pad, column 64 == ones

typedef __attribute__((ext_vector_type(16))) __bf16          v16bf;
typedef __attribute__((ext_vector_type(16))) unsigned short  v16u;
typedef __attribute__((ext_vector_type(8)))  float           v8f;

__device__ __forceinline__ unsigned short f32_to_bf16(float f) {
    __bf16 b = (__bf16)f;                       // hardware cvt, RNE
    return __builtin_bit_cast(unsigned short, b);
}

__device__ __forceinline__ float lane_bcast(float v, int srcLane) {
    return __int_as_float(__builtin_amdgcn_ds_bpermute(srcLane << 2, __float_as_int(v)));
}

// Per-lane async copy of 16 bytes global -> LDS (CDNA5 GLOBAL_LOAD_ASYNC_TO_LDS,
// tracked by ASYNCcnt). ldsOff is the byte offset within the wave's LDS space.
__device__ __forceinline__ void async_g2l_b128(unsigned ldsOff, const void* g) {
    unsigned long long ga = (unsigned long long)g;
    asm volatile("global_load_async_to_lds_b128 %0, %1, off"
                 :: "v"(ldsOff), "v"(ga) : "memory");
}
__device__ __forceinline__ void wait_async0() {
    asm volatile("s_wait_asynccnt 0x0" ::: "memory");
}

// Load a 16x32 bf16 fragment from an LDS matrix whose "outer" dim (M for A,
// N for B) is the row index and K is contiguous (stride = ld halfwords).
// Matches the ISA 16-bit A-matrix 16x32 layout:
//   lane&15 = outer index, kbase = (lane>=16)*8,
//   halves 0..7 -> K = kofs+kbase+0..7, halves 8..15 -> K = kofs+16+kbase+0..7
__device__ __forceinline__ v16u frag16x32(const unsigned short* base, int ld,
                                          int outerBase, int kofs) {
    int lane = threadIdx.x & 31;
    int o  = outerBase + (lane & 15);
    int kb = kofs + ((lane >> 4) << 3);
    const unsigned short* p = base + o * ld + kb;
    union { uint4 q[2]; v16u v; } t;
    t.q[0] = *(const uint4*)(p);
    t.q[1] = *(const uint4*)(p + 16);
    return t.v;
}

__device__ __forceinline__ v8f wmma_bf16(v16u a, v16u b, v8f c) {
    return __builtin_amdgcn_wmma_f32_16x16x32_bf16(
        false, __builtin_bit_cast(v16bf, a),
        false, __builtin_bit_cast(v16bf, b),
        (short)0, c, false, false);
}

// ---------------------------------------------------------------------------
// Kernel 1: pad/convert v -> bf16 v' (H,L,DP), col 64 = 1.0, cols 65..79 = 0
// ---------------------------------------------------------------------------
__global__ void vpad_kernel(const float* __restrict__ v, unsigned short* __restrict__ vP) {
    long idx = (long)blockIdx.x * blockDim.x + threadIdx.x;
    if (idx >= (long)H_ * L_ * DP_) return;
    int n   = (int)(idx % DP_);
    long hl = idx / DP_;
    unsigned short o;
    if (n < D_)        o = f32_to_bf16(v[hl * D_ + n]);
    else if (n == D_)  o = 0x3F80;   // bf16 1.0
    else               o = 0;
    vP[idx] = o;
}

// ---------------------------------------------------------------------------
// Kernel 2: phi(x) = exp(W^T x' - ||x'||^2/2) / sqrt(R), x' = 0.125 * x
// One block per (h, chunk): (128x64) @ (64x128) via WMMA. 256 thr = 8 waves.
// ---------------------------------------------------------------------------
__global__ void phi_kernel(const float* __restrict__ x, const float* __restrict__ W,
                           unsigned short* __restrict__ phi) {
    extern __shared__ char smem[];
    unsigned short* xs = (unsigned short*)smem;       // 128*64 bf16, row-major A
    unsigned short* Wt = xs + C_ * D_;                // 128*64 bf16, W^T (B col-major)
    float* sq   = (float*)(Wt + R_ * D_);             // 128
    float* part = sq + C_;                            // 256

    int blk = blockIdx.x;                 // h*NC + c
    int h = blk >> 5, c = blk & 31;
    int t = threadIdx.x;

    // load x chunk, scale, square-sum, convert to bf16
    {
        int row = t >> 1, half = t & 1;
        const float* src = x + ((long)h * L_ + (long)c * C_ + row) * D_ + half * 32;
        float s = 0.f;
        #pragma unroll
        for (int i = 0; i < 32; ++i) {
            float vv = src[i] * 0.125f;
            s += vv * vv;
            xs[row * D_ + half * 32 + i] = f32_to_bf16(vv);
        }
        part[t] = s;
    }
    // load W (D x R f32) transposed -> Wt[r*64 + d] bf16
    for (int i = 0; i < 32; ++i) {
        int idx = t * 32 + i;              // covers 64*128 = 8192
        int d = idx >> 7, r = idx & 127;
        Wt[r * D_ + d] = f32_to_bf16(W[idx]);
    }
    __syncthreads();
    if (t < C_) sq[t] = 0.5f * (part[2 * t] + part[2 * t + 1]);
    __syncthreads();

    int wave = t >> 5, lane = t & 31;
    int hi = lane >> 4, ln = lane & 15;
    int mbase = wave * 16;
    v16u a0 = frag16x32(xs, D_, mbase, 0);
    v16u a1 = frag16x32(xs, D_, mbase, 32);
    for (int nt = 0; nt < 8; ++nt) {
        v8f acc = {};
        acc = wmma_bf16(a0, frag16x32(Wt, D_, nt * 16, 0),  acc);
        acc = wmma_bf16(a1, frag16x32(Wt, D_, nt * 16, 32), acc);
        #pragma unroll
        for (int j = 0; j < 8; ++j) {
            int row = mbase + j + hi * 8;
            float ph = __expf(acc[j] - sq[row]) * 0.08838834764831845f; // 1/sqrt(128)
            long l = (long)h * L_ + (long)c * C_ + row;
            phi[l * R_ + nt * 16 + ln] = f32_to_bf16(ph);
        }
    }
}

// ---------------------------------------------------------------------------
// Kernel 3: per-chunk skv' = pk_c^T @ v'_c  -> (R=128 x DP=80) f32, K=128
// ---------------------------------------------------------------------------
__global__ void skv_kernel(const unsigned short* __restrict__ phiK,
                           const unsigned short* __restrict__ vP,
                           float* __restrict__ skv) {
    extern __shared__ char smem[];
    unsigned short* pkT = (unsigned short*)smem;  // [r*128 + j]  (A row-major in K)
    unsigned short* vT  = pkT + R_ * C_;          // [n*128 + j]  (B col-major)

    int blk = blockIdx.x;
    int h = blk >> 5, c = blk & 31;
    int t = threadIdx.x;

    long pbase = ((long)h * L_ + (long)c * C_) * R_;
    for (int i = 0; i < 64; ++i) {                // 128*128 transpose
        int idx = t + i * 256;
        int j = idx >> 7, r = idx & 127;
        pkT[r * C_ + j] = phiK[pbase + idx];
    }
    long vbase = ((long)h * L_ + (long)c * C_) * DP_;
    for (int i = 0; i < 40; ++i) {                // 128*80 transpose
        int idx = t + i * 256;
        int j = idx / DP_, n = idx % DP_;
        vT[n * C_ + j] = vP[vbase + idx];
    }
    __syncthreads();

    int wave = t >> 5, lane = t & 31;
    int hi = lane >> 4, ln = lane & 15;
    int rbase = wave * 16;
    long obase = (long)blk * R_ * DP_;
    for (int nt = 0; nt < 5; ++nt) {
        v8f acc = {};
        #pragma unroll
        for (int ks = 0; ks < 4; ++ks)
            acc = wmma_bf16(frag16x32(pkT, C_, rbase, ks * 32),
                            frag16x32(vT,  C_, nt * 16, ks * 32), acc);
        #pragma unroll
        for (int j = 0; j < 8; ++j) {
            int r = rbase + j + hi * 8;
            skv[obase + (long)r * DP_ + nt * 16 + ln] = acc[j];
        }
    }
}

// ---------------------------------------------------------------------------
// Kernel 4: exclusive prefix over chunks, in place. Each thread owns one
// (h, r, n) strip -> deterministic; skv is fully rewritten each call.
// ---------------------------------------------------------------------------
__global__ void prefix_kernel(float* __restrict__ skv) {
    int idx = blockIdx.x * blockDim.x + threadIdx.x;   // over H*R*DP
    if (idx >= H_ * R_ * DP_) return;
    int h  = idx / (R_ * DP_);
    int rn = idx % (R_ * DP_);
    float run = 0.f;
    for (int c = 0; c < NC_; ++c) {
        long off = ((long)(h * NC_ + c)) * (R_ * DP_) + rn;
        float cur = skv[off];
        skv[off] = run;
        run += cur;
    }
}

// ---------------------------------------------------------------------------
// Kernel 5: per (h,chunk) output.
// Phase 1: S = pq @ pk^T, strict-lower mask, -> bf16 LDS.
// Phase 2: out' = pq @ skv_prev + S @ v'   (128 x 80); col 64 = denominator.
// pq/pk chunk copies use CDNA5 async global->LDS (ASYNCcnt), overlapped with
// the transposed vT/skvT fills done through VGPRs.
// ---------------------------------------------------------------------------
__global__ void out_kernel(const unsigned short* __restrict__ phiQ,
                           const unsigned short* __restrict__ phiK,
                           const unsigned short* __restrict__ vP,
                           const float* __restrict__ skv,
                           float* __restrict__ out) {
    extern __shared__ char smem[];
    unsigned short* pq   = (unsigned short*)smem;    // 128x128 row-major (A)
    unsigned short* pk   = pq + C_ * R_;             // 128x128 row-major == (pk^T) col-major (B)
    unsigned short* skvT = pk + C_ * R_;             // [n*128 + r], n<80 (B)
    unsigned short* vT   = skvT + DP_ * C_;          // [n*128 + s], n<80 (B)
    unsigned short* aM   = vT + DP_ * C_;            // 128x128 masked S, row-major (A)

    int blk = blockIdx.x;
    int h = blk >> 5, c = blk & 31;
    int t = threadIdx.x;

    long pbase = ((long)h * L_ + (long)c * C_) * R_;
    // ---- async straight copies of pq, pk chunks (32 KB each) ----
    {
        // dynamic LDS begins right after static LDS (none here)
        unsigned ldsBase = __builtin_amdgcn_groupstaticsize();
        unsigned pqOff = ldsBase;                          // byte offset of pq
        unsigned pkOff = ldsBase + C_ * R_ * 2;            // byte offset of pk
        const unsigned short* gq = phiQ + pbase;
        const unsigned short* gk = phiK + pbase;
        #pragma unroll
        for (int i = 0; i < 8; ++i) {
            unsigned e = (unsigned)(t + i * 256) * 8u;     // halfword index, 8 per 16B
            async_g2l_b128(pqOff + e * 2u, gq + e);
            async_g2l_b128(pkOff + e * 2u, gk + e);
        }
    }
    // ---- transposed fills through VGPRs (2-byte granularity) ----
    long vbase = ((long)h * L_ + (long)c * C_) * DP_;
    for (int i = 0; i < 40; ++i) {
        int idx = t + i * 256;
        int j = idx / DP_, n = idx % DP_;
        vT[n * C_ + j] = vP[vbase + idx];
    }
    long sbase = (long)blk * R_ * DP_;
    for (int i = 0; i < 40; ++i) {
        int idx = t + i * 256;
        int r = idx / DP_, n = idx % DP_;
        skvT[n * C_ + r] = f32_to_bf16(skv[sbase + idx]);
    }
    wait_async0();          // ASYNCcnt == 0: async LDS writes landed
    __syncthreads();

    int wave = t >> 5, lane = t & 31;
    int hi = lane >> 4, ln = lane & 15;
    int mbase = wave * 16;

    // ---- Phase 1: masked S -> aM (bf16) ----
    {
        v16u a[4];
        #pragma unroll
        for (int ks = 0; ks < 4; ++ks) a[ks] = frag16x32(pq, R_, mbase, ks * 32);
        for (int nt = 0; nt < 8; ++nt) {
            int nbase = nt * 16;
            if (nbase >= mbase + 16) {             // fully above strict-lower region
                #pragma unroll
                for (int j = 0; j < 8; ++j)
                    aM[(mbase + j + hi * 8) * R_ + nbase + ln] = 0;
                continue;
            }
            v8f acc = {};
            #pragma unroll
            for (int ks = 0; ks < 4; ++ks)
                acc = wmma_bf16(a[ks], frag16x32(pk, R_, nbase, ks * 32), acc);
            int s = nbase + ln;
            #pragma unroll
            for (int j = 0; j < 8; ++j) {
                int row = mbase + j + hi * 8;
                float val = (s < row) ? acc[j] : 0.f;   // strictly causal (j < l)
                aM[row * R_ + nbase + ln] = f32_to_bf16(val);
            }
        }
    }
    __syncthreads();

    // ---- Phase 2: out' = pq @ skv_prev + S @ v' ----
    {
        v16u aq[4], ai[4];
        #pragma unroll
        for (int ks = 0; ks < 4; ++ks) {
            aq[ks] = frag16x32(pq, R_, mbase, ks * 32);
            ai[ks] = frag16x32(aM, R_, mbase, ks * 32);
        }
        v8f acc[5];
        for (int nt = 0; nt < 5; ++nt) {
            v8f a = {};
            #pragma unroll
            for (int ks = 0; ks < 4; ++ks) {
                a = wmma_bf16(aq[ks], frag16x32(skvT, C_, nt * 16, ks * 32), a);
                a = wmma_bf16(ai[ks], frag16x32(vT,   C_, nt * 16, ks * 32), a);
            }
            acc[nt] = a;
        }
        // denominator = column 64 (ones column) of acc[4]; lives in ln==0 lanes
        long ob = (long)h * L_ + (long)c * C_;
        #pragma unroll
        for (int j = 0; j < 8; ++j) {
            float den  = lane_bcast(acc[4][j], lane & 16) + 1e-10f;
            float rden = 1.0f / den;
            int row = mbase + j + hi * 8;
            long rowoff = (ob + row) * D_;
            #pragma unroll
            for (int nt = 0; nt < 4; ++nt)
                out[rowoff + nt * 16 + ln] = acc[nt][j] * rden;
        }
    }
}

// ---------------------------------------------------------------------------
// launch
// ---------------------------------------------------------------------------
extern "C" void kernel_launch(void* const* d_in, const int* in_sizes, int n_in,
                              void* d_out, int out_size, void* d_ws, size_t ws_size,
                              hipStream_t stream) {
    // inputs: T (unused), k, q, v, W
    const float* k = (const float*)d_in[1];
    const float* q = (const float*)d_in[2];
    const float* v = (const float*)d_in[3];
    const float* W = (const float*)d_in[4];
    float* out = (float*)d_out;

    // workspace partition
    unsigned short* phiQ = (unsigned short*)d_ws;                      // H*L*R bf16
    unsigned short* phiK = phiQ + (size_t)H_ * L_ * R_;                // H*L*R bf16
    unsigned short* vP   = phiK + (size_t)H_ * L_ * R_;                // H*L*DP bf16
    float*          skv  = (float*)(vP + (size_t)H_ * L_ * DP_);      // H*NC*R*DP f32

    const int SMEM_PHI = (C_ * D_ + R_ * D_) * 2 + (C_ + 256) * 4;              // ~34 KB
    const int SMEM_SKV = (R_ * C_ + DP_ * C_) * 2;                              // 52 KB
    const int SMEM_OUT = (C_ * R_ * 3 + DP_ * C_ * 2) * 2;                      // 136 KB

    long nvp = (long)H_ * L_ * DP_;
    vpad_kernel  <<<(int)((nvp + 255) / 256),      256, 0,        stream>>>(v, vP);
    phi_kernel   <<<H_ * NC_,                      256, SMEM_PHI, stream>>>(q, W, phiQ);
    phi_kernel   <<<H_ * NC_,                      256, SMEM_PHI, stream>>>(k, W, phiK);
    skv_kernel   <<<H_ * NC_,                      256, SMEM_SKV, stream>>>(phiK, vP, skv);
    prefix_kernel<<<(H_ * R_ * DP_ + 255) / 256,   256, 0,        stream>>>(skv);
    out_kernel   <<<H_ * NC_,                      256, SMEM_OUT, stream>>>(phiQ, phiK, vP, skv, out);
}
